// ProbAttention_12326556140039
// MI455X (gfx1250) — matrix operand order repaired and checked by reading, compiled, not run on hardware
//
#include <hip/hip_runtime.h>

// ---------------------------------------------------------------------------
// ProbAttention for MI455X (gfx1250), bf16 WMMA path, async-LDS double buffer.
//   B=8, L=S=4096, D=1024, H=16, E=64, U=40
// ---------------------------------------------------------------------------

typedef __attribute__((ext_vector_type(16))) __bf16 bf16x16;
typedef __attribute__((ext_vector_type(8)))  float  f32x8;

union FragB16 {
  bf16x16 v;
  uint4   q[2];
  unsigned short h[16];
};

__device__ __forceinline__ unsigned short f32_to_bf16_rne(float f) {
  union { float f; unsigned u; } c; c.f = f;
  unsigned u = c.u;
  u += 0x7FFFu + ((u >> 16) & 1u);   // round-to-nearest-even
  return (unsigned short)(u >> 16);
}

// Pack two f32 -> packed bf16x2 (one VALU op on gfx1250 if available).
__device__ __forceinline__ unsigned pack2_bf16(float lo, float hi) {
#if __has_builtin(__builtin_amdgcn_cvt_pk_bf16_f32)
  auto r = __builtin_amdgcn_cvt_pk_bf16_f32(lo, hi);
  unsigned u;
  __builtin_memcpy(&u, &r, 4);
  return u;
#else
  return (unsigned)f32_to_bf16_rne(lo) | ((unsigned)f32_to_bf16_rne(hi) << 16);
#endif
}

__device__ __forceinline__ void zero_acc(f32x8& a) {
  #pragma unroll
  for (int j = 0; j < 8; ++j) a[j] = 0.0f;
}

__device__ __forceinline__ void zero_frag(FragB16& f) {
  f.q[0] = uint4{0u, 0u, 0u, 0u};
  f.q[1] = uint4{0u, 0u, 0u, 0u};
}

// CDNA5 async global -> LDS copy (ASYNCcnt path). 16 bytes per lane.
// IOFFSET is added to BOTH the LDS and global addresses by hardware, so a
// single base register pair serves a whole row of copies (no address VALU).
template <int OFF>
__device__ __forceinline__ void async_load_b128_off(unsigned lds_off, const void* gaddr) {
  asm volatile("global_load_async_to_lds_b128 %0, %1, off offset:%2"
               :: "v"(lds_off), "v"(gaddr), "i"(OFF) : "memory");
}
__device__ __forceinline__ void wait_async0() {
  asm volatile("s_wait_asynccnt 0" ::: "memory");
}

// ---------------------------------------------------------------------------
// f32 -> bf16 bulk convert (weights)
// ---------------------------------------------------------------------------
__global__ void cvt_f32_to_bf16(const float* __restrict__ src,
                                unsigned short* __restrict__ dst, int n) {
  int i = blockIdx.x * blockDim.x + threadIdx.x;   // n/2 threads, 2 elems each
  int j = i * 2;
  if (j < n) {
    unsigned p = pack2_bf16(src[j], src[j + 1]);
    *(unsigned*)(dst + j) = p;
  }
}

// ---------------------------------------------------------------------------
// Gather the U=40 sampled query rows per batch: Qs_in[b*U+u, d]
// ---------------------------------------------------------------------------
__global__ void gather_qs(const float* __restrict__ q,
                          const int* __restrict__ idx,
                          float* __restrict__ out) {
  int t = blockIdx.x * blockDim.x + threadIdx.x;     // 320*1024 threads
  if (t >= 320 * 1024) return;
  int d  = t & 1023;
  int bu = t >> 10;
  int b = bu / 40, u = bu % 40;
  out[t] = q[((size_t)b * 4096 + (size_t)idx[u]) * 1024 + d];
}

// ---------------------------------------------------------------------------
// Generic GEMM: C[M,1024] = A_f32[M,1024] x W_bf16[1024,1024]^T + bias
// REQUIRES: M a multiple of 64 (all call sites satisfy this; no edge guards).
// out_mode 0: f32 row-major [M,1024]
// out_mode 1: bf16 row-major [M,1024]
// out_mode 2: bf16 transposed per-head [b,h,e,s]  (requires M == B*S = 32768)
// Block: 256 threads (8 waves). Block tile 64(M) x 256(N), K-step 64
// (two 32-deep WMMA sub-steps per stage), double-buffered LDS; W tile fetched
// with global_load_async_to_lds_b128 using immediate offsets.
// Wave (wm in 0..1, wn in 0..3) owns 32(M) x 64(N) = 2x4 WMMA tiles.
// ---------------------------------------------------------------------------
__global__ __launch_bounds__(256) void gemm_xwT(
    const float* __restrict__ A, const unsigned short* __restrict__ W,
    const float* __restrict__ bias, void* __restrict__ Cout,
    int M, int out_mode)
{
  __shared__ __align__(16) unsigned short As[2][64 * 64];    // 16 KB
  __shared__ __align__(16) unsigned short Bs[2][256 * 64];   // 64 KB

  const int tid  = threadIdx.x;
  const int lane = tid & 31;
  const int wid  = tid >> 5;       // 0..7
  const int wm   = wid >> 2;       // 0..1 : 32-row half
  const int wn   = wid & 3;        // 0..3 : 64-col quarter
  const int hi   = lane >> 4;      // lane-group (K half selector)
  const int ln   = lane & 15;

  const int m0 = blockIdx.x * 64;
  const int n0 = blockIdx.y * 256;

  f32x8 acc[2][4];
  #pragma unroll
  for (int mt = 0; mt < 2; ++mt)
    #pragma unroll
    for (int nt = 0; nt < 4; ++nt) zero_acc(acc[mt][nt]);

  // cooperative A-staging coordinates (64 rows x 64 k, f32 -> bf16)
  const int arow = tid >> 2;          // 0..63
  const int acol = (tid & 3) * 16;    // 0,16,32,48  (16 f32 each)

  // staging lambdas -----------------------------------------------------
  auto stageA = [&](int k0, int buf) {
    const float* p = A + (size_t)(m0 + arow) * 1024 + k0 + acol;
    float t0[16];
    #pragma unroll
    for (int i = 0; i < 16; ++i) t0[i] = p[i];
    uint4 pk0, pk1;
    pk0.x = pack2_bf16(t0[0],  t0[1]);
    pk0.y = pack2_bf16(t0[2],  t0[3]);
    pk0.z = pack2_bf16(t0[4],  t0[5]);
    pk0.w = pack2_bf16(t0[6],  t0[7]);
    pk1.x = pack2_bf16(t0[8],  t0[9]);
    pk1.y = pack2_bf16(t0[10], t0[11]);
    pk1.z = pack2_bf16(t0[12], t0[13]);
    pk1.w = pack2_bf16(t0[14], t0[15]);
    uint4* q = (uint4*)(&As[buf][arow * 64 + acol]);
    q[0] = pk0; q[1] = pk1;
  };
  auto stageW = [&](int k0, int buf) {
    // one W row (64 bf16 = 128B) per thread, via 8 async b128 copies that
    // share one base address pair and use hardware IOFFSET for both sides.
    const unsigned short* wp = W + (size_t)(n0 + tid) * 1024 + k0;
    unsigned lb = (unsigned)(size_t)&Bs[buf][tid * 64];
    async_load_b128_off<  0>(lb, wp);
    async_load_b128_off< 16>(lb, wp);
    async_load_b128_off< 32>(lb, wp);
    async_load_b128_off< 48>(lb, wp);
    async_load_b128_off< 64>(lb, wp);
    async_load_b128_off< 80>(lb, wp);
    async_load_b128_off< 96>(lb, wp);
    async_load_b128_off<112>(lb, wp);
  };
  // ---------------------------------------------------------------------

  stageA(0, 0);
  stageW(0, 0);
  wait_async0();
  __syncthreads();

  const int NITER = 1024 / 64;
  for (int it = 0; it < NITER; ++it) {
    const int cur = it & 1;
    const int k0  = it * 64;

    if (it + 1 < NITER) {           // stage next tile while computing
      stageA(k0 + 64, cur ^ 1);
      stageW(k0 + 64, cur ^ 1);
    }

    // two 32-deep WMMA sub-steps from the current buffers
    #pragma unroll
    for (int ks = 0; ks < 2; ++ks) {
      FragB16 bfrag[4];
      #pragma unroll
      for (int nt = 0; nt < 4; ++nt) {
        const int r = wn * 64 + nt * 16 + ln;             // B column (N)
        const uint4* p = (const uint4*)(&Bs[cur][r * 64 + ks * 32 + hi * 8]);
        bfrag[nt].q[0] = p[0]; bfrag[nt].q[1] = p[2];
      }
      FragB16 afrag[2];
      #pragma unroll
      for (int mt = 0; mt < 2; ++mt) {
        const int r = wm * 32 + mt * 16 + ln;             // A row (M)
        afrag[mt].q[0] = *(const uint4*)(&As[cur][r * 64 + ks * 32 + hi * 8]);
        afrag[mt].q[1] = *(const uint4*)(&As[cur][r * 64 + ks * 32 + 16 + hi * 8]);
      }
      #pragma unroll
      for (int mt = 0; mt < 2; ++mt)
        #pragma unroll
        for (int nt = 0; nt < 4; ++nt)
          acc[mt][nt] = __builtin_amdgcn_wmma_f32_16x16x32_bf16(
              false, afrag[mt].v, false, bfrag[nt].v,
              (short)0, acc[mt][nt], false, false);
    }

    wait_async0();      // next-buffer async copies complete
    __syncthreads();    // everyone done reading cur / writing next
  }

  // ---- epilogue: bias + store (M multiple of 64 -> no row guard) ----
  #pragma unroll
  for (int mt = 0; mt < 2; ++mt) {
    #pragma unroll
    for (int nt = 0; nt < 4; ++nt) {
      #pragma unroll
      for (int j = 0; j < 8; ++j) {
        const int row = m0 + wm * 32 + mt * 16 + hi * 8 + j;
        const int col = n0 + wn * 64 + nt * 16 + ln;
        const float v = acc[mt][nt][j] + bias[col];
        if (out_mode == 0) {
          ((float*)Cout)[(size_t)row * 1024 + col] = v;
        } else if (out_mode == 1) {
          ((unsigned short*)Cout)[(size_t)row * 1024 + col] = f32_to_bf16_rne(v);
        } else {
          // V transposed per head: [b, h, e, s]  (row = b*4096+s, col = h*64+e)
          const int b = row >> 12, s = row & 4095;
          const int h = col >> 6,  e = col & 63;
          ((unsigned short*)Cout)[((((size_t)(b * 16 + h)) * 64 + e) << 12) + s] =
              f32_to_bf16_rne(v);
        }
      }
    }
  }
}

// ---------------------------------------------------------------------------
// scores[b,h,u,s] = scale * sum_e Qs[b,u,h,e] * Kp[b,s,h,e]
// Block 256 = 8 waves; each wave: one 16-wide s strip, 3 u-tiles (U=40 -> 48).
// grid.x = (B*H) * (S/128) = 128 * 32 = 4096
// ---------------------------------------------------------------------------
__global__ __launch_bounds__(256) void scores_kernel(
    const unsigned short* __restrict__ Qs,   // [B,U,1024] bf16
    const unsigned short* __restrict__ Kp,   // [B,S,1024] bf16
    float* __restrict__ attn)                // [B,H,U,S]  f32
{
  const int lane = threadIdx.x & 31, wid = threadIdx.x >> 5;
  const int hi = lane >> 4, ln = lane & 15;
  const int bh = blockIdx.x >> 5;            // 0..127
  const int sc = blockIdx.x & 31;            // 0..31
  const int b = bh >> 4, h = bh & 15;
  const int s0 = sc * 128 + wid * 16;

  f32x8 acc[3];
  #pragma unroll
  for (int ut = 0; ut < 3; ++ut) zero_acc(acc[ut]);

  #pragma unroll
  for (int estep = 0; estep < 2; ++estep) {
    FragB16 bf_;
    {
      const uint4* p = (const uint4*)(Kp + ((size_t)(b * 4096 + s0 + ln)) * 1024
                                         + h * 64 + estep * 32 + hi * 8);
      bf_.q[0] = p[0]; bf_.q[1] = p[2];
    }
    #pragma unroll
    for (int ut = 0; ut < 3; ++ut) {
      FragB16 af;
      const int u = ut * 16 + ln;
      if (u < 40) {
        const unsigned short* base =
            Qs + ((size_t)(b * 40 + u)) * 1024 + h * 64 + estep * 32;
        af.q[0] = *(const uint4*)(base + hi * 8);
        af.q[1] = *(const uint4*)(base + 16 + hi * 8);
      } else {
        zero_frag(af);
      }
      acc[ut] = __builtin_amdgcn_wmma_f32_16x16x32_bf16(
          false, af.v, false, bf_.v, (short)0, acc[ut], false, false);
    }
  }

  const float scale = 0.03125f;  // 1024^-0.5
  #pragma unroll
  for (int ut = 0; ut < 3; ++ut)
    #pragma unroll
    for (int j = 0; j < 8; ++j) {
      const int u = ut * 16 + hi * 8 + j;
      if (u < 40)
        attn[(((size_t)bh * 40 + u) << 12) + s0 + ln] = acc[ut][j] * scale;
    }
}

// ---------------------------------------------------------------------------
// Row softmax over S=4096; normalizes attn (f32, in place) and emits bf16 copy.
// One block (256 threads) per row; 5120 rows.
// ---------------------------------------------------------------------------
__global__ __launch_bounds__(256) void softmax_kernel(
    float* __restrict__ attn, unsigned short* __restrict__ attn_bf)
{
  __shared__ float red[256];
  const int row = blockIdx.x;
  float* p = attn + ((size_t)row << 12);
  const int t = threadIdx.x;

  float x[16];
  float mx = -1e30f;
  #pragma unroll
  for (int i = 0; i < 16; ++i) { x[i] = p[t + i * 256]; mx = fmaxf(mx, x[i]); }
  red[t] = mx; __syncthreads();
  for (int off = 128; off > 0; off >>= 1) {
    if (t < off) red[t] = fmaxf(red[t], red[t + off]);
    __syncthreads();
  }
  mx = red[0]; __syncthreads();

  float sum = 0.0f;
  #pragma unroll
  for (int i = 0; i < 16; ++i) { x[i] = __expf(x[i] - mx); sum += x[i]; }
  red[t] = sum; __syncthreads();
  for (int off = 128; off > 0; off >>= 1) {
    if (t < off) red[t] += red[t + off];
    __syncthreads();
  }
  const float inv = 1.0f / red[0];

  unsigned short* pb = attn_bf + ((size_t)row << 12);
  #pragma unroll
  for (int i = 0; i < 16; ++i) {
    const float v = x[i] * inv;
    p[t + i * 256]  = v;
    pb[t + i * 256] = f32_to_bf16_rne(v);
  }
}

// ---------------------------------------------------------------------------
// Partial PV: for s-chunk c (512 wide):
//   parts[c][ (b*U+u)*1024 + h*64+e ] = sum_{s in chunk} attn_bf * Vt
// grid (128, 8), block 128 (4 waves, wave w owns e-tile w*16..w*16+15).
// ---------------------------------------------------------------------------
__global__ __launch_bounds__(128) void pv_partial_kernel(
    const unsigned short* __restrict__ attn_bf,  // [B,H,U,S] bf16
    const unsigned short* __restrict__ Vt,       // [B,H,E,S] bf16
    float* __restrict__ parts)                   // [8][320*1024] f32
{
  const int lane = threadIdx.x & 31, w = threadIdx.x >> 5;
  const int hi = lane >> 4, ln = lane & 15;
  const int bh = blockIdx.x;
  const int ck = blockIdx.y;                    // 0..7
  const int b = bh >> 4, h = bh & 15;

  f32x8 acc[3];
  #pragma unroll
  for (int ut = 0; ut < 3; ++ut) zero_acc(acc[ut]);

  const unsigned short* vbase = Vt + (((size_t)bh * 64) << 12);
  const int sbeg = ck * 512, send = sbeg + 512;

  for (int s0 = sbeg; s0 < send; s0 += 32) {
    FragB16 bf_;
    {
      const uint4* p = (const uint4*)(vbase + (((size_t)(w * 16 + ln)) << 12)
                                            + s0 + hi * 8);
      bf_.q[0] = p[0]; bf_.q[1] = p[2];
    }
    #pragma unroll
    for (int ut = 0; ut < 3; ++ut) {
      const int u = ut * 16 + ln;
      FragB16 af;
      if (u < 40) {
        const unsigned short* ab = attn_bf + (((size_t)bh * 40 + u) << 12) + s0;
        af.q[0] = *(const uint4*)(ab + hi * 8);
        af.q[1] = *(const uint4*)(ab + 16 + hi * 8);
      } else {
        zero_frag(af);
      }
      acc[ut] = __builtin_amdgcn_wmma_f32_16x16x32_bf16(
          false, af.v, false, bf_.v, (short)0, acc[ut], false, false);
    }
  }

  float* dst = parts + (size_t)ck * 320 * 1024;
  #pragma unroll
  for (int ut = 0; ut < 3; ++ut)
    #pragma unroll
    for (int j = 0; j < 8; ++j) {
      const int u = ut * 16 + hi * 8 + j;
      if (u < 40)
        dst[((size_t)(b * 40 + u)) * 1024 + h * 64 + w * 16 + ln] = acc[ut][j];
    }
}

__global__ void reduce_outh(const float* __restrict__ parts,
                            float* __restrict__ outh) {
  int t = blockIdx.x * blockDim.x + threadIdx.x;
  if (t >= 320 * 1024) return;
  float s = 0.0f;
  #pragma unroll
  for (int c = 0; c < 8; ++c) s += parts[(size_t)c * 320 * 1024 + t];
  outh[t] = s;
}

// ---------------------------------------------------------------------------
// Workspace layout (bytes). Total ~198 MB.
// ---------------------------------------------------------------------------
static constexpr size_t WS_WQB   = 0;                                    // bf16 1024x1024
static constexpr size_t WS_WKB   = WS_WQB  + (size_t)1024 * 1024 * 2;
static constexpr size_t WS_WVB   = WS_WKB  + (size_t)1024 * 1024 * 2;
static constexpr size_t WS_WOB   = WS_WVB  + (size_t)1024 * 1024 * 2;
static constexpr size_t WS_QSIN  = WS_WOB  + (size_t)1024 * 1024 * 2;    // f32 320x1024
static constexpr size_t WS_QSB   = WS_QSIN + (size_t)320 * 1024 * 4;     // bf16 320x1024
static constexpr size_t WS_OUTH  = WS_QSB  + (size_t)320 * 1024 * 2;     // f32 320x1024
static constexpr size_t WS_KP    = WS_OUTH + (size_t)320 * 1024 * 4;     // bf16 [B,S,1024]
static constexpr size_t WS_VT    = WS_KP   + (size_t)8 * 4096 * 1024 * 2;// bf16 [B,H,E,S]
static constexpr size_t WS_ATTNB = WS_VT   + (size_t)8 * 4096 * 1024 * 2;// bf16 [B,H,U,S]
static constexpr size_t WS_PARTS = WS_ATTNB+ (size_t)8 * 16 * 40 * 4096 * 2; // f32 [8][320*1024]

extern "C" void kernel_launch(void* const* d_in, const int* in_sizes, int n_in,
                              void* d_out, int out_size, void* d_ws, size_t ws_size,
                              hipStream_t stream) {
  (void)in_sizes; (void)n_in; (void)out_size; (void)ws_size;

  const float* queries = (const float*)d_in[0];
  const float* keys    = (const float*)d_in[1];
  const float* values  = (const float*)d_in[2];
  const float* Wq = (const float*)d_in[3];
  const float* bq = (const float*)d_in[4];
  const float* Wk = (const float*)d_in[5];
  const float* bk = (const float*)d_in[6];
  const float* Wv = (const float*)d_in[7];
  const float* bv = (const float*)d_in[8];
  const float* Wo = (const float*)d_in[9];
  const float* bo = (const float*)d_in[10];
  const int* sidx = (const int*)d_in[11];

  char* ws = (char*)d_ws;
  unsigned short* Wqb   = (unsigned short*)(ws + WS_WQB);
  unsigned short* Wkb   = (unsigned short*)(ws + WS_WKB);
  unsigned short* Wvb   = (unsigned short*)(ws + WS_WVB);
  unsigned short* Wob   = (unsigned short*)(ws + WS_WOB);
  float*          QsIn  = (float*)(ws + WS_QSIN);
  unsigned short* Qsb   = (unsigned short*)(ws + WS_QSB);
  float*          outh  = (float*)(ws + WS_OUTH);
  unsigned short* Kpb   = (unsigned short*)(ws + WS_KP);
  unsigned short* Vtb   = (unsigned short*)(ws + WS_VT);
  unsigned short* attnb = (unsigned short*)(ws + WS_ATTNB);
  float*          parts = (float*)(ws + WS_PARTS);

  float* out  = (float*)d_out;              // [B,U,1024] = 327,680 f32
  float* attn = out + (size_t)320 * 1024;   // [B,H,U,S]  = 20,971,520 f32

  // 1) weights -> bf16 (2 elems/thread)
  cvt_f32_to_bf16<<<2048, 256, 0, stream>>>(Wq, Wqb, 1 << 20);
  cvt_f32_to_bf16<<<2048, 256, 0, stream>>>(Wk, Wkb, 1 << 20);
  cvt_f32_to_bf16<<<2048, 256, 0, stream>>>(Wv, Wvb, 1 << 20);
  cvt_f32_to_bf16<<<2048, 256, 0, stream>>>(Wo, Wob, 1 << 20);

  // 2) gather sampled query rows (only these need the Q projection)
  gather_qs<<<1280, 256, 0, stream>>>(queries, sidx, QsIn);

  // 3) projections (WMMA bf16 GEMMs, 64x256 tiles, K-step 64)
  gemm_xwT<<<dim3(5, 4),   256, 0, stream>>>(QsIn,   Wqb, bq, Qsb, 320,   1);
  gemm_xwT<<<dim3(512, 4), 256, 0, stream>>>(keys,   Wkb, bk, Kpb, 32768, 1);
  gemm_xwT<<<dim3(512, 4), 256, 0, stream>>>(values, Wvb, bv, Vtb, 32768, 2);

  // 4) scores -> attn (f32, part of the output)
  scores_kernel<<<4096, 256, 0, stream>>>(Qsb, Kpb, attn);

  // 5) softmax rows (normalize in place + bf16 copy for PV)
  softmax_kernel<<<5120, 256, 0, stream>>>(attn, attnb);

  // 6) attn @ V per head, split over 8 s-chunks for occupancy, then reduce
  pv_partial_kernel<<<dim3(128, 8), 128, 0, stream>>>(attnb, Vtb, parts);
  reduce_outh<<<1280, 256, 0, stream>>>(parts, outh);

  // 7) output projection
  gemm_xwT<<<dim3(5, 4), 256, 0, stream>>>(outh, Wob, bo, out, 320, 0);
}
